// Transformer_89438398972616
// MI455X (gfx1250) — compile-verified, matrix operand order checked by reading, not run on hardware
//
#include <hip/hip_runtime.h>
#include <cstddef>
#include <cstdint>

// ---------------------------------------------------------------------------
// Transformer forward (eval) for MI455X / gfx1250.
// Dense GEMMs on v_wmma_f32_16x16x32_bf16 (fp32 -> bf16 in LDS tile loaders,
// fp32 accumulate). Double-buffered LDS tiles: global loads for tile k+1 are
// issued before the WMMA work on tile k so load latency hides under compute.
// bf16 packing uses v_cvt_pk_bf16_f32 directly (one op per f32 pair).
// Attention softmax / LN are VALU kernels (<2% of FLOPs).
// Shapes: M=2048 (B*S), K in {512,2048}, N in {512,2048,32000};
// all divisible by the 128x128x32 block tile.
// ---------------------------------------------------------------------------

#define Dm   512
#define Hh   8
#define DKc  64
#define DFFc 2048
#define Vc   32000
#define NLc  6
#define Bc   4
#define Sc   512
#define BS   (Bc * Sc)   // 2048 tokens

typedef __attribute__((ext_vector_type(16))) __bf16 v16bf;
typedef __attribute__((ext_vector_type(8)))  float  v8f;

union Frag16 { v16bf v; uint4 q[2]; unsigned short u[16]; };

// pack two f32 into one u32 of bf16x2 with a single native instruction
__device__ __forceinline__ unsigned int pkbf2(float a, float b) {
  unsigned int r;
  asm("v_cvt_pk_bf16_f32 %0, %1, %2" : "=v"(r) : "v"(a), "v"(b));
  return r;
}

// ---------------------------------------------------------------------------
// GEMM: C[M,N] = A[M,K] @ W[N,K]^T + bias (optional ReLU). torch Linear layout.
// Block = 256 threads (8 waves). Tile BM=128 x BN=128 x BK=32, double-buffered.
// Wave grid 4x2; each wave computes a 32x64 patch = 2x4 WMMA 16x16 tiles.
// ---------------------------------------------------------------------------
#define BM  128
#define BN  128
#define BKt 32
#define LDT 40   // padded LDS row stride in ushorts (32 + 8 pad)

// fetch one 128x32 fp32 tile: 4 float4 per thread, all loads issued together
__device__ __forceinline__ void fetch_tile(float4* r, const float* __restrict__ base,
                                           int ld, int t) {
  #pragma unroll
  for (int i = 0; i < 4; ++i) {
    int idx = t + (i << 8);                 // 0..1023
    r[i] = *(const float4*)(base + (size_t)(idx >> 3) * ld + ((idx & 7) << 2));
  }
}

// convert + store one staged tile into LDS as bf16 (2x u32 -> b64 merge)
__device__ __forceinline__ void stage_tile(const float4* r, unsigned short* dst, int t) {
  #pragma unroll
  for (int i = 0; i < 4; ++i) {
    int idx = t + (i << 8);
    int row = idx >> 3;
    int c4  = (idx & 7) << 2;
    unsigned int* p = (unsigned int*)&dst[row * LDT + c4];
    p[0] = pkbf2(r[i].x, r[i].y);
    p[1] = pkbf2(r[i].z, r[i].w);
  }
}

__global__ void __launch_bounds__(256)
gemm_bias_wmma(const float* __restrict__ A, const float* __restrict__ W,
               const float* __restrict__ bias, float* __restrict__ C,
               int M, int N, int K, int relu)
{
  __shared__ __align__(16) unsigned short As[2][BM * LDT];
  __shared__ __align__(16) unsigned short Bs[2][BN * LDT];

  const int t    = threadIdx.x;
  const int wave = t >> 5;
  const int lane = t & 31;
  const int wr   = wave & 3;     // wave row (0..3)  -> 32-row strip
  const int wc   = wave >> 2;    // wave col (0..1)  -> 64-col strip
  const int lh   = lane >> 4;    // lane half (K-group select per ISA layout)
  const int lr   = lane & 15;    // row/col within 16
  const int m0   = blockIdx.x * BM;
  const int n0   = blockIdx.y * BN;

  const float* Abase = A + (size_t)m0 * K;
  const float* Wbase = W + (size_t)n0 * K;

  v8f acc[2][4] = {};
  float4 ra[4], rb[4];

  // ---- prologue: stage tile 0 ----
  fetch_tile(ra, Abase, K, t);
  fetch_tile(rb, Wbase, K, t);
  stage_tile(ra, As[0], t);
  stage_tile(rb, Bs[0], t);
  __syncthreads();

  const int NK = K / BKt;
  for (int kt = 0; kt < NK; ++kt) {
    const int cur = kt & 1;

    // issue global loads for tile k+1 before computing tile k
    if (kt + 1 < NK) {
      const size_t k0 = (size_t)(kt + 1) * BKt;
      fetch_tile(ra, Abase + k0, K, t);
      fetch_tile(rb, Wbase + k0, K, t);
    }
    // software prefetch two tiles ahead (global_prefetch_b8 path)
    if (kt + 2 < NK) {
      __builtin_prefetch(Abase + (size_t)(t & 127) * K + (size_t)(kt + 2) * BKt, 0, 1);
      __builtin_prefetch(Wbase + (size_t)(t & 127) * K + (size_t)(kt + 2) * BKt, 0, 1);
    }

    // ---- gather fragments (contiguous 16B chunks -> ds_load_b128) ----
    Frag16 af[2], bfr[4];
    #pragma unroll
    for (int i = 0; i < 2; ++i) {
      const unsigned short* p = &As[cur][(wr * 32 + i * 16 + lr) * LDT + (lh << 3)];
      af[i].q[0] = *(const uint4*)p;
      af[i].q[1] = *(const uint4*)(p + 16);
    }
    #pragma unroll
    for (int j = 0; j < 4; ++j) {
      const unsigned short* p = &Bs[cur][(wc * 64 + j * 16 + lr) * LDT + (lh << 3)];
      bfr[j].q[0] = *(const uint4*)p;
      bfr[j].q[1] = *(const uint4*)(p + 16);
    }

    #pragma unroll
    for (int i = 0; i < 2; ++i)
      #pragma unroll
      for (int j = 0; j < 4; ++j)
        acc[i][j] = __builtin_amdgcn_wmma_f32_16x16x32_bf16(
            false, af[i].v, false, bfr[j].v, (short)0, acc[i][j], false, false);

    // stage next tile into the other LDS buffer; one barrier per iteration
    if (kt + 1 < NK) {
      stage_tile(ra, As[cur ^ 1], t);
      stage_tile(rb, Bs[cur ^ 1], t);
      __syncthreads();
    }
  }

  // ---- epilogue: C/D layout -> lane lr = N col, VGPR r -> M row r + 8*lh ----
  #pragma unroll
  for (int j = 0; j < 4; ++j) {
    const int n  = n0 + wc * 64 + j * 16 + lr;
    const float bv = bias ? bias[n] : 0.f;
    #pragma unroll
    for (int i = 0; i < 2; ++i) {
      #pragma unroll
      for (int r = 0; r < 8; ++r) {
        const int m = m0 + wr * 32 + i * 16 + r + (lh << 3);
        float val = acc[i][j][r] + bv;
        if (relu) val = fmaxf(val, 0.f);
        C[(size_t)m * N + n] = val;
      }
    }
  }
}

// ---------------------------------------------------------------------------
// Attention: one block per (b, h, q). S=512 keys, d_k=64. q pre-scaled 1/8.
// ---------------------------------------------------------------------------
__global__ void __launch_bounds__(128)
attn_kernel(const float* __restrict__ Q, const float* __restrict__ Km,
            const float* __restrict__ V, float* __restrict__ O, int causal)
{
  __shared__ float qv[DKc];
  __shared__ float sc[Sc];
  __shared__ float red[128];

  const int q = blockIdx.x, h = blockIdx.y, b = blockIdx.z;
  const int t = threadIdx.x;
  const size_t base = (size_t)b * Sc * Dm + (size_t)h * DKc;

  if (t < DKc) qv[t] = Q[base + (size_t)q * Dm + t] * 0.125f;  // 1/sqrt(64)
  __syncthreads();

  float lmax = -3.4e38f;
  for (int k = t; k < Sc; k += 128) {
    float s;
    if (causal && (k > q)) {
      s = -1e9f;                                 // matches reference NEG
    } else {
      const float* kp = Km + base + (size_t)k * Dm;
      s = 0.f;
      #pragma unroll 16
      for (int d = 0; d < DKc; ++d) s += qv[d] * kp[d];
    }
    sc[k] = s;
    lmax = fmaxf(lmax, s);
  }
  red[t] = lmax; __syncthreads();
  for (int o = 64; o > 0; o >>= 1) { if (t < o) red[t] = fmaxf(red[t], red[t + o]); __syncthreads(); }
  const float mx = red[0]; __syncthreads();

  float lsum = 0.f;
  for (int k = t; k < Sc; k += 128) { float e = __expf(sc[k] - mx); sc[k] = e; lsum += e; }
  red[t] = lsum; __syncthreads();
  for (int o = 64; o > 0; o >>= 1) { if (t < o) red[t] += red[t + o]; __syncthreads(); }
  const float inv = 1.f / red[0]; __syncthreads();

  // O[q, h*64 + d] = sum_k p[k] * V[k, h*64 + d]; split keys across 2 halves
  const int d  = t & (DKc - 1);
  const int hh = t >> 6;
  const float* vp = V + base + d;
  float oacc = 0.f;
  for (int k = hh * 256; k < hh * 256 + 256; ++k) oacc += sc[k] * vp[(size_t)k * Dm];
  red[t] = oacc; __syncthreads();
  if (t < DKc) O[base + (size_t)q * Dm + t] = (red[t] + red[t + DKc]) * inv;
}

// ---------------------------------------------------------------------------
// Fused residual-add + LayerNorm (biased variance, eps 1e-5). One block/row.
// ---------------------------------------------------------------------------
__global__ void __launch_bounds__(256)
add_ln_kernel(const float* x, const float* a,
              const float* __restrict__ g, const float* __restrict__ beta,
              float* out)
{
  __shared__ float red[256];
  const int row = blockIdx.x;
  const int t   = threadIdx.x;
  const size_t off = (size_t)row * Dm;

  float v0 = x[off + t]       + a[off + t];
  float v1 = x[off + t + 256] + a[off + t + 256];
  red[t] = v0 + v1; __syncthreads();
  for (int o = 128; o > 0; o >>= 1) { if (t < o) red[t] += red[t + o]; __syncthreads(); }
  const float mean = red[0] * (1.0f / (float)Dm);
  __syncthreads();

  const float d0 = v0 - mean, d1 = v1 - mean;
  red[t] = d0 * d0 + d1 * d1; __syncthreads();
  for (int o = 128; o > 0; o >>= 1) { if (t < o) red[t] += red[t + o]; __syncthreads(); }
  const float rstd = rsqrtf(red[0] * (1.0f / (float)Dm) + 1e-5f);

  out[off + t]       = g[t]       * d0 * rstd + beta[t];
  out[off + t + 256] = g[t + 256] * d1 * rstd + beta[t + 256];
}

// ---------------------------------------------------------------------------
// Embedding lookup + on-device sinusoidal positional encoding.
// ---------------------------------------------------------------------------
__global__ void __launch_bounds__(256)
embed_pe_kernel(const int* __restrict__ tok, const float* __restrict__ emb,
                float* __restrict__ out)
{
  const int idx = blockIdx.x * 256 + threadIdx.x;   // [0, BS*Dm)
  const int d   = idx & (Dm - 1);
  const int bs  = idx >> 9;                         // / Dm (512)
  const int s   = bs & (Sc - 1);
  const int token = tok[bs];
  const int pair  = d >> 1;
  const float freq = __expf(-9.2103403719761836f * ((float)(pair << 1) / (float)Dm));
  const float ang  = (float)s * freq;
  const float pe   = (d & 1) ? __cosf(ang) : __sinf(ang);
  out[idx] = emb[(size_t)token * Dm + d] + pe;
}

__global__ void __launch_bounds__(256)
copy_kernel(const float* __restrict__ in, float* __restrict__ out, int n)
{
  int i = blockIdx.x * 256 + threadIdx.x;
  if (i < n) out[i] = in[i];
}

// ---------------------------------------------------------------------------
// Host orchestration. Flattened param order (recursive insertion order):
//  [0]=src [1]=tgt [2]=src_emb [3]=tgt_emb
//  enc layer l (base 4+16l):  wq bq wk bk wv bv wo bo  w1 b1 w2 b2  ln1g ln1b ln2g ln2b
//  dec layer l (base 100+26l): self(wq..bo) cross(wq..bo) w1 b1 w2 b2 ln1g ln1b ln2g ln2b ln3g ln3b
//  [256]=final_w [257]=final_b
// ---------------------------------------------------------------------------
extern "C" void kernel_launch(void* const* d_in, const int* in_sizes, int n_in,
                              void* d_out, int out_size, void* d_ws, size_t ws_size,
                              hipStream_t stream)
{
  (void)in_sizes; (void)n_in; (void)out_size; (void)ws_size;

  const int*   src = (const int*)d_in[0];
  const int*   tgt = (const int*)d_in[1];
  const float* src_emb = (const float*)d_in[2];
  const float* tgt_emb = (const float*)d_in[3];
  const float* const* P = (const float* const*)d_in;

  float* ws = (float*)d_ws;
  const size_t NTOK = (size_t)BS * Dm;          // 1,048,576 floats
  float* wsX = ws;                 // running activations (x / y)
  float* wsA = wsX + NTOK;         // sublayer output (pre-LN residual add)
  float* wsQ = wsA + NTOK;
  float* wsK = wsQ + NTOK;
  float* wsV = wsK + NTOK;
  float* wsO = wsV + NTOK;         // concat attention heads
  float* wsE = wsO + NTOK;         // encoder output snapshot
  float* wsH = wsE + NTOK;         // FFN hidden [BS, DFF]

  auto gemm = [&](const float* Aa, const float* Ww, const float* bb, float* Cc,
                  int M, int N, int K, int relu) {
    dim3 g(M / BM, N / BN);
    gemm_bias_wmma<<<g, 256, 0, stream>>>(Aa, Ww, bb, Cc, M, N, K, relu);
  };
  const dim3 agrid(Sc, Hh, Bc);
  const int elemBlocks = (BS * Dm) / 256;       // 4096

  // ---------------- encoder ----------------
  embed_pe_kernel<<<elemBlocks, 256, 0, stream>>>(src, src_emb, wsX);
  for (int l = 0; l < NLc; ++l) {
    const int b0 = 4 + 16 * l;
    gemm(wsX, P[b0 + 0], P[b0 + 1], wsQ, BS, Dm, Dm, 0);
    gemm(wsX, P[b0 + 2], P[b0 + 3], wsK, BS, Dm, Dm, 0);
    gemm(wsX, P[b0 + 4], P[b0 + 5], wsV, BS, Dm, Dm, 0);
    attn_kernel<<<agrid, 128, 0, stream>>>(wsQ, wsK, wsV, wsO, 0);
    gemm(wsO, P[b0 + 6], P[b0 + 7], wsA, BS, Dm, Dm, 0);
    add_ln_kernel<<<BS, 256, 0, stream>>>(wsX, wsA, P[b0 + 12], P[b0 + 13], wsX);
    gemm(wsX, P[b0 + 8], P[b0 + 9],  wsH, BS, DFFc, Dm, 1);     // ReLU
    gemm(wsH, P[b0 + 10], P[b0 + 11], wsA, BS, Dm, DFFc, 0);
    add_ln_kernel<<<BS, 256, 0, stream>>>(wsX, wsA, P[b0 + 14], P[b0 + 15], wsX);
  }
  copy_kernel<<<elemBlocks, 256, 0, stream>>>(wsX, wsE, BS * Dm);

  // ---------------- decoder ----------------
  embed_pe_kernel<<<elemBlocks, 256, 0, stream>>>(tgt, tgt_emb, wsX);
  for (int l = 0; l < NLc; ++l) {
    const int b0 = 100 + 26 * l;
    // masked self-attention
    gemm(wsX, P[b0 + 0], P[b0 + 1], wsQ, BS, Dm, Dm, 0);
    gemm(wsX, P[b0 + 2], P[b0 + 3], wsK, BS, Dm, Dm, 0);
    gemm(wsX, P[b0 + 4], P[b0 + 5], wsV, BS, Dm, Dm, 0);
    attn_kernel<<<agrid, 128, 0, stream>>>(wsQ, wsK, wsV, wsO, 1);
    gemm(wsO, P[b0 + 6], P[b0 + 7], wsA, BS, Dm, Dm, 0);
    add_ln_kernel<<<BS, 256, 0, stream>>>(wsX, wsA, P[b0 + 20], P[b0 + 21], wsX);
    // cross-attention (K/V from encoder output)
    gemm(wsX, P[b0 + 8],  P[b0 + 9],  wsQ, BS, Dm, Dm, 0);
    gemm(wsE, P[b0 + 10], P[b0 + 11], wsK, BS, Dm, Dm, 0);
    gemm(wsE, P[b0 + 12], P[b0 + 13], wsV, BS, Dm, Dm, 0);
    attn_kernel<<<agrid, 128, 0, stream>>>(wsQ, wsK, wsV, wsO, 0);
    gemm(wsO, P[b0 + 14], P[b0 + 15], wsA, BS, Dm, Dm, 0);
    add_ln_kernel<<<BS, 256, 0, stream>>>(wsX, wsA, P[b0 + 22], P[b0 + 23], wsX);
    // FFN
    gemm(wsX, P[b0 + 16], P[b0 + 17], wsH, BS, DFFc, Dm, 1);    // ReLU
    gemm(wsH, P[b0 + 18], P[b0 + 19], wsA, BS, Dm, DFFc, 0);
    add_ln_kernel<<<BS, 256, 0, stream>>>(wsX, wsA, P[b0 + 24], P[b0 + 25], wsX);
  }

  // -------- final vocab projection (65MB weight stays resident in 192MB L2) --------
  gemm(wsX, P[256], P[257], (float*)d_out, BS, Vc, Dm, 0);
}
